// TimeMixingTriton_34110630265513
// MI455X (gfx1250) — compile-verified, matrix operand order checked by reading, not run on hardware
//
#include <hip/hip_runtime.h>

// ---------------------------------------------------------------------------
// RWKV-7 time mixing for MI455X (gfx1250, wave32, WMMA + async LDS copies)
// B=8, T=2048, C=1024, H=16, N=64
// ---------------------------------------------------------------------------

typedef __attribute__((ext_vector_type(16))) __bf16 v16bf;
typedef __attribute__((ext_vector_type(4)))  __bf16 v4bf;
typedef __attribute__((ext_vector_type(8)))  float  v8f;

#define Bc 8
#define Tc 2048
#define Cc 1024
#define Hc 16
#define Nc 64

#define BM 128
#define BN 128
#define BK 32
#define LDSW 40           // 32 + 8 pad (16B-aligned rows, breaks bank conflicts)

constexpr int Mtok   = Bc * Tc;     // 16384
constexpr int Kc     = Cc;          // 1024
constexpr int KSTEPS = Kc / BK;     // 32

// ---------------------------------------------------------------------------
// fp32 -> bf16 conversion, 4 elements/thread (packed cvt + b64 stores)
// ---------------------------------------------------------------------------
__global__ void cvt_bf16_kernel(const float4* __restrict__ in,
                                v4bf* __restrict__ out, int n4) {
    int i = blockIdx.x * blockDim.x + threadIdx.x;
    if (i < n4) {
        float4 f = in[i];
        v4bf o;
        o[0] = (__bf16)f.x; o[1] = (__bf16)f.y;
        o[2] = (__bf16)f.z; o[3] = (__bf16)f.w;
        out[i] = o;
    }
}

// ---------------------------------------------------------------------------
// Async global->LDS copy of 16B per lane (ASYNCcnt-tracked, CDNA5 §15.18)
// lds_addr = low 32 bits of the generic pointer (LDS aperture offset).
// ---------------------------------------------------------------------------
__device__ inline void async_copy_b128(unsigned int lds_addr, const void* gptr) {
    asm volatile("global_load_async_to_lds_b128 %0, %1, off"
                 :: "v"(lds_addr),
                    "v"((unsigned long long)(uintptr_t)gptr)
                 : "memory");
}

// ---------------------------------------------------------------------------
// Load one 16x32 bf16 WMMA fragment from an LDS row (ISA 7.12.2 layout):
// lane[3:0] = row/col, lane[4] = K-half; two contiguous 16B chunks.
// ---------------------------------------------------------------------------
__device__ inline v16bf load_frag(const __bf16* row, int khalf) {
    union { v16bf v; uint4 u[2]; } f;
    f.u[0] = *(const uint4*)(row + khalf * 8);
    f.u[1] = *(const uint4*)(row + 16 + khalf * 8);
    return f.v;
}

// ---------------------------------------------------------------------------
// C[16384,1024] = A[16384,1024](bf16) x W[1024,1024]^T (bf16, row-major [N][K])
// MODE 0: f32   MODE 1: bf16 sigmoid   MODE 2: f32 exp(-exp)
// 256 threads = 8 waves; 128x128 tile; wave patch 32x64 = 2x4 WMMA accums.
// Double-buffered LDS fed by async b128 copies.
// ---------------------------------------------------------------------------
template <int MODE>
__global__ __launch_bounds__(256) void gemm_wmma_kernel(
    const __bf16* __restrict__ A, const __bf16* __restrict__ W,
    float* outF, __bf16* outB) {
    __shared__ __bf16 As[2][BM][LDSW];
    __shared__ __bf16 Bs[2][BN][LDSW];

    const int tid  = threadIdx.x;
    const int wave = tid >> 5;
    const int lane = tid & 31;
    const int wm   = (wave >> 1) * 32;   // 0,32,64,96
    const int wn   = (wave & 1) * 64;    // 0,64
    const int bM   = blockIdx.y * BM;
    const int bN   = blockIdx.x * BN;
    const int khalf = lane >> 4;
    const int lr    = lane & 15;

    // this thread's two 16B chunks per tile: chunk = tid + c*256
    const int row0  = tid >> 2;               // c=0: rows 0..63
    const int row1  = (tid + 256) >> 2;       // c=1: rows 64..127
    const int col8a = (tid & 3) * 8;

    auto issue_stage = [&](int s, int buf) {
        const int k0 = s * BK;
        async_copy_b128((unsigned int)(uintptr_t)&As[buf][row0][col8a],
                        &A[(size_t)(bM + row0) * Kc + k0 + col8a]);
        async_copy_b128((unsigned int)(uintptr_t)&Bs[buf][row0][col8a],
                        &W[(size_t)(bN + row0) * Kc + k0 + col8a]);
        async_copy_b128((unsigned int)(uintptr_t)&As[buf][row1][col8a],
                        &A[(size_t)(bM + row1) * Kc + k0 + col8a]);
        async_copy_b128((unsigned int)(uintptr_t)&Bs[buf][row1][col8a],
                        &W[(size_t)(bN + row1) * Kc + k0 + col8a]);
    };

    v8f acc[2][4] = {};

    issue_stage(0, 0);
    for (int s = 0; s < KSTEPS; ++s) {
        const int cur = s & 1;
        if (s + 1 < KSTEPS) {
            issue_stage(s + 1, cur ^ 1);          // prefetch next tile
            asm volatile("s_wait_asynccnt 0x4" ::: "memory");  // cur tile landed
        } else {
            asm volatile("s_wait_asynccnt 0x0" ::: "memory");
        }
        __syncthreads();                          // all waves' tiles visible

        v16bf af[2], bfr[4];
#pragma unroll
        for (int tm = 0; tm < 2; ++tm)
            af[tm] = load_frag(&As[cur][wm + tm * 16 + lr][0], khalf);
#pragma unroll
        for (int tn = 0; tn < 4; ++tn)
            bfr[tn] = load_frag(&Bs[cur][wn + tn * 16 + lr][0], khalf);

#pragma unroll
        for (int tm = 0; tm < 2; ++tm)
#pragma unroll
            for (int tn = 0; tn < 4; ++tn)
                acc[tm][tn] = __builtin_amdgcn_wmma_f32_16x16x32_bf16(
                    false, af[tm], false, bfr[tn], (short)0, acc[tm][tn],
                    false, false);
        __syncthreads();          // done reading cur before it is refilled
    }

    // Epilogue: lane[3:0]=col, VGPR i -> row = i + 8*lane[4]; row stride 1024
#pragma unroll
    for (int tm = 0; tm < 2; ++tm)
#pragma unroll
        for (int tn = 0; tn < 4; ++tn) {
            const int col  = bN + wn + tn * 16 + lr;
            const size_t b = ((size_t)(bM + wm + tm * 16 + khalf * 8) << 10) + col;
            if (MODE == 0) {
                float* p = outF + b;
#pragma unroll
                for (int i = 0; i < 8; ++i) p[(size_t)i << 10] = acc[tm][tn][i];
            } else if (MODE == 1) {
                __bf16* p = outB + b;
#pragma unroll
                for (int i = 0; i < 8; ++i)
                    p[(size_t)i << 10] =
                        (__bf16)(1.0f / (1.0f + __expf(-acc[tm][tn][i])));
            } else {
                float* p = outF + b;
#pragma unroll
                for (int i = 0; i < 8; ++i)
                    p[(size_t)i << 10] = __expf(-__expf(acc[tm][tn][i]));
            }
        }
}

// ---------------------------------------------------------------------------
// WKV7 scan. One block per (b,h); 64 threads; thread i owns state row i.
// S[i][j] = S[i][j]*w[i] + k[i]*v[j]; y_i = sum_j S[i][j]*r[j].
// y written in-place over the r buffer.
// ---------------------------------------------------------------------------
__global__ __launch_bounds__(64) void wkv7_scan_kernel(
    const float* r, const float* w, const float* k, const float* v,
    const float* state_in, float* y, float* state_out) {
    const int bh = blockIdx.x;
    const int b  = bh / Hc;
    const int h  = bh - b * Hc;
    const int i  = threadIdx.x;

    float s[Nc];
    const size_t sbase = ((size_t)bh * Nc + i) * Nc;
#pragma unroll
    for (int j = 0; j < Nc; ++j) s[j] = state_in[sbase + j];

    __shared__ float rr[Nc];
    __shared__ float vv[Nc];

    for (int t = 0; t < Tc; ++t) {
        const size_t base = (((size_t)b * Tc + t) * Hc + h) * Nc;
        float ri = r[base + i];
        float wi = w[base + i];
        float ki = k[base + i];
        float vi = v[base + i];
        __syncthreads();
        rr[i] = ri;
        vv[i] = vi;
        __syncthreads();
        float acc = 0.0f;
#pragma unroll
        for (int j = 0; j < Nc; ++j) {
            s[j] = fmaf(s[j], wi, ki * vv[j]);
            acc  = fmaf(s[j], rr[j], acc);
        }
        y[base + i] = acc;
    }

#pragma unroll
    for (int j = 0; j < Nc; ++j) state_out[sbase + j] = s[j];
}

// ---------------------------------------------------------------------------
// LayerNorm over C=1024 then * sigmoid-gate -> bf16. One block per token.
// ---------------------------------------------------------------------------
__global__ __launch_bounds__(256) void ln_gate_kernel(
    const float* __restrict__ y, const __bf16* __restrict__ g,
    const float* __restrict__ gamma, const float* __restrict__ beta,
    __bf16* __restrict__ out) {
    const int row = blockIdx.x;
    const int tid = threadIdx.x;
    const float* yr = y + ((size_t)row << 10);

    float v[4], sum = 0.0f, sq = 0.0f;
#pragma unroll
    for (int i = 0; i < 4; ++i) {
        v[i] = yr[tid + 256 * i];
        sum += v[i];
        sq  += v[i] * v[i];
    }
    __shared__ float s1[256], s2[256];
    s1[tid] = sum;
    s2[tid] = sq;
    __syncthreads();
    for (int off = 128; off > 0; off >>= 1) {
        if (tid < off) {
            s1[tid] += s1[tid + off];
            s2[tid] += s2[tid + off];
        }
        __syncthreads();
    }
    const float mean = s1[0] * (1.0f / Cc);
    const float var  = s2[0] * (1.0f / Cc) - mean * mean;
    const float rstd = rsqrtf(var + 1e-5f);

#pragma unroll
    for (int i = 0; i < 4; ++i) {
        int   c   = tid + 256 * i;
        float val = (v[i] - mean) * rstd * gamma[c] + beta[c];
        out[((size_t)row << 10) + c] =
            (__bf16)(val * (float)g[((size_t)row << 10) + c]);
    }
}

// ---------------------------------------------------------------------------
// Launcher
// ---------------------------------------------------------------------------
extern "C" void kernel_launch(void* const* d_in, const int* in_sizes, int n_in,
                              void* d_out, int out_size, void* d_ws,
                              size_t ws_size, hipStream_t stream) {
    const float* x     = (const float*)d_in[0];
    const float* state = (const float*)d_in[1];
    const float* W_r   = (const float*)d_in[2];
    const float* W_k   = (const float*)d_in[3];
    const float* W_v   = (const float*)d_in[4];
    const float* W_o   = (const float*)d_in[5];
    const float* W_g   = (const float*)d_in[6];
    const float* W_w   = (const float*)d_in[7];
    const float* gamma = (const float*)d_in[8];
    const float* beta  = (const float*)d_in[9];
    float* out = (float*)d_out;

    const size_t MC = (size_t)Mtok * Cc;   // 16,777,216
    const size_t WW = (size_t)Cc * Cc;     // 1,048,576

    char*  ws  = (char*)d_ws;
    size_t off = 0;
    auto alloc = [&](size_t bytes) -> void* {
        void* p = ws + off;
        off += (bytes + 255) & ~(size_t)255;
        return p;
    };
    __bf16* xb   = (__bf16*)alloc(MC * 2);     // reused later as yg (bf16)
    __bf16* Wrb  = (__bf16*)alloc(WW * 2);
    __bf16* Wkb  = (__bf16*)alloc(WW * 2);
    __bf16* Wvb  = (__bf16*)alloc(WW * 2);
    __bf16* Wob  = (__bf16*)alloc(WW * 2);
    __bf16* Wgb  = (__bf16*)alloc(WW * 2);
    __bf16* Wwb  = (__bf16*)alloc(WW * 2);
    float*  rbuf = (float*)alloc(MC * 4);      // reused as y by the scan
    float*  kbuf = (float*)alloc(MC * 4);
    float*  vbuf = (float*)alloc(MC * 4);
    float*  wbuf = (float*)alloc(MC * 4);
    __bf16* gbuf = (__bf16*)alloc(MC * 2);

    // 1) convert to bf16 (vectorized x4)
    cvt_bf16_kernel<<<(unsigned)(MC / 4 / 256), 256, 0, stream>>>(
        (const float4*)x, (v4bf*)xb, (int)(MC / 4));
    const float* Wsrc[6] = {W_r, W_k, W_v, W_o, W_g, W_w};
    __bf16*      Wdst[6] = {Wrb, Wkb, Wvb, Wob, Wgb, Wwb};
    for (int i = 0; i < 6; ++i)
        cvt_bf16_kernel<<<(unsigned)(WW / 4 / 256), 256, 0, stream>>>(
            (const float4*)Wsrc[i], (v4bf*)Wdst[i], (int)(WW / 4));

    // 2) projections (WMMA bf16 GEMMs)
    dim3 grid(Cc / BN, Mtok / BM);
    gemm_wmma_kernel<0><<<grid, 256, 0, stream>>>(xb, Wrb, rbuf, nullptr);
    gemm_wmma_kernel<0><<<grid, 256, 0, stream>>>(xb, Wkb, kbuf, nullptr);
    gemm_wmma_kernel<0><<<grid, 256, 0, stream>>>(xb, Wvb, vbuf, nullptr);
    gemm_wmma_kernel<2><<<grid, 256, 0, stream>>>(xb, Wwb, wbuf, nullptr);
    gemm_wmma_kernel<1><<<grid, 256, 0, stream>>>(xb, Wgb, nullptr, gbuf);

    // 3) WKV7 scan: y overwrites rbuf; final state -> tail of d_out
    wkv7_scan_kernel<<<Bc * Hc, Nc, 0, stream>>>(rbuf, wbuf, kbuf, vbuf, state,
                                                 rbuf, out + MC);

    // 4) LN + gate -> bf16 (reuse xb region)
    __bf16* ygb = xb;
    ln_gate_kernel<<<Mtok, 256, 0, stream>>>(rbuf, gbuf, gamma, beta, ygb);

    // 5) output projection -> d_out
    gemm_wmma_kernel<0><<<grid, 256, 0, stream>>>(ygb, Wob, out, nullptr);
}
